// EventTransformer_76785425318263
// MI455X (gfx1250) — matrix operand order repaired and checked by reading, not compiled, not run on hardware
//
#include <hip/hip_runtime.h>
#include <hip/hip_bf16.h>
#include <math.h>

// ---- problem constants (match reference) ----
#define D_MODEL 256
#define NHEAD   8
#define HD      32
#define NLAYER  4
#define DFF     1024
#define T_DOMS  32768
#define BATCH   64
#define SMAX    768
#define BS      (BATCH*SMAX)   // 49152 padded rows

typedef __attribute__((ext_vector_type(16))) _Float16 v16h;
typedef __attribute__((ext_vector_type(8)))  float    v8f;

struct h2 { _Float16 x, y; };

__device__ inline v8f wmma16(v16h a, v16h b, v8f c) {
  // D = A(16x32 f16) * B(32x16 f16) + C(16x16 f32)
  return __builtin_amdgcn_wmma_f32_16x16x32_f16(false, a, false, b, (short)0, c, false, false);
}

// A-fragment K offset per ISA table: lanes 0-15 hold K {0..7,16..23}, lanes 16-31 hold K {8..15,24..31}
__device__ inline int a_k0(int v, int hsel) {
  return ((v < 4) ? 2 * v : 16 + 2 * (v - 4)) + 8 * hsel;
}

// ------------------------------------------------------------------
// packing helpers
// ------------------------------------------------------------------
__global__ void count_kernel(const int* __restrict__ ev, int* __restrict__ counts) {
  int i = blockIdx.x * 256 + threadIdx.x;
  if (i < T_DOMS) atomicAdd(&counts[ev[i]], 1);
}

__global__ void scan_kernel(const int* __restrict__ counts, int* __restrict__ offsets) {
  if (threadIdx.x == 0) {
    int acc = 0;
    for (int b = 0; b < BATCH; ++b) { offsets[b] = acc; acc += counts[b]; }
  }
}

__global__ void pos_kernel(const int* __restrict__ ev, const int* __restrict__ offsets,
                           int* __restrict__ pos) {
  int i = blockIdx.x * 256 + threadIdx.x;
  if (i < T_DOMS) pos[i] = i - offsets[ev[i]];
}

// ------------------------------------------------------------------
// geometry MLP: hidden = gelu(geo/500 @ W1^T + b1)   (T x 128)
// ------------------------------------------------------------------
__global__ __launch_bounds__(128)
void geo_hidden_kernel(const float* __restrict__ geom, const int* __restrict__ dom_ids,
                       const float* __restrict__ w1, const float* __restrict__ b1,
                       float* __restrict__ geoh) {
  int t = blockIdx.x, j = threadIdx.x;
  int sid = dom_ids[t];
  float g0 = geom[sid * 3 + 0] * 0.002f;
  float g1 = geom[sid * 3 + 1] * 0.002f;
  float g2 = geom[sid * 3 + 2] * 0.002f;
  float z = w1[j * 3 + 0] * g0 + w1[j * 3 + 1] * g1 + w1[j * 3 + 2] * g2 + b1[j];
  geoh[(size_t)t * 128 + j] = 0.5f * z * (1.0f + erff(z * 0.70710678118f)); // exact gelu
}

// feat = emb + hidden @ W2^T + b2; scatter into padded x and set valid
__global__ __launch_bounds__(256)
void geo_scatter_kernel(const float* __restrict__ emb, const float* __restrict__ geoh,
                        const float* __restrict__ w2, const float* __restrict__ b2,
                        const int* __restrict__ ev, const int* __restrict__ pos,
                        float* __restrict__ x, int* __restrict__ valid) {
  int t = blockIdx.x, d = threadIdx.x;
  __shared__ float hs[128];
  if (d < 128) hs[d] = geoh[(size_t)t * 128 + d];
  __syncthreads();
  float acc = b2[d];
  const float* wr = w2 + (size_t)d * 128;
  #pragma unroll 4
  for (int k = 0; k < 128; ++k) acc += wr[k] * hs[k];
  float f = emb[(size_t)t * D_MODEL + d] + acc;
  int e = ev[t], p = pos[t];
  if (p < SMAX) {
    int row = e * SMAX + p;
    x[(size_t)row * D_MODEL + d] = f;
    if (d == 0) valid[row] = 1;
  }
}

// ------------------------------------------------------------------
// weight convert: W (N x K, f32) -> WT (K x N, f16)
// ------------------------------------------------------------------
__global__ void wcast_kernel(const float* __restrict__ W, _Float16* __restrict__ WT,
                             int N, int K) {
  int i = blockIdx.x * 256 + threadIdx.x;
  if (i < N * K) {
    int n = i / K, k = i % K;
    WT[(size_t)k * N + n] = (_Float16)W[i];
  }
}

// ------------------------------------------------------------------
// layernorm: f32 row -> f16 row
// ------------------------------------------------------------------
__global__ __launch_bounds__(256)
void ln_kernel(const float* __restrict__ x, const float* __restrict__ w,
               const float* __restrict__ b, _Float16* __restrict__ out) {
  __shared__ float red[256];
  int row = blockIdx.x, tid = threadIdx.x;
  float v = x[(size_t)row * D_MODEL + tid];
  red[tid] = v; __syncthreads();
  for (int s = 128; s > 0; s >>= 1) { if (tid < s) red[tid] += red[tid + s]; __syncthreads(); }
  float mean = red[0] * (1.0f / D_MODEL); __syncthreads();
  float d = v - mean;
  red[tid] = d * d; __syncthreads();
  for (int s = 128; s > 0; s >>= 1) { if (tid < s) red[tid] += red[tid + s]; __syncthreads(); }
  float r = rsqrtf(red[0] * (1.0f / D_MODEL) + 1e-5f);
  out[(size_t)row * D_MODEL + tid] = (_Float16)(d * r * w[tid] + b[tid]);
}

// ------------------------------------------------------------------
// WMMA GEMM: C[M,N] = A[M,K](f16) * BT[K,N](f16) + bias, specialized epilogue.
// block = 4 waves, each wave computes a 32x64 tile; block tile = 128x64
//   MODE 0: out f16 = acc + bias                       (QKV)
//   MODE 1: out f16 = relu(acc + bias)                 (FFN1)
//   MODE 2: out f32 = acc + bias + Cf[row,col] in place (out-proj / FFN2 residual)
// ------------------------------------------------------------------
template <int MODE>
__global__ __launch_bounds__(128)
void gemm_wmma_kernel(const _Float16* __restrict__ A, const _Float16* __restrict__ BT,
                      const float* __restrict__ bias, float* __restrict__ Cf,
                      _Float16* __restrict__ Ch, int N, int K) {
  const int lane = threadIdx.x & 31;
  const int wave = threadIdx.x >> 5;
  const int m_base = blockIdx.x * 128 + wave * 32;
  const int n_base = blockIdx.y * 64;
  const int hsel = lane >> 4;
  const int lm = lane & 15;

  v8f acc[2][4];
  #pragma unroll
  for (int r = 0; r < 2; ++r)
    #pragma unroll
    for (int j = 0; j < 4; ++j) acc[r][j] = (v8f){};

  for (int kc = 0; kc < K; kc += 32) {
    v16h a[2];
    #pragma unroll
    for (int r = 0; r < 2; ++r) {
      const _Float16* ap = A + (size_t)(m_base + r * 16 + lm) * K + kc;
      #pragma unroll
      for (int v = 0; v < 8; ++v) {
        h2 p = *(const h2*)(ap + a_k0(v, hsel));
        a[r][2 * v] = p.x; a[r][2 * v + 1] = p.y;
      }
    }
    v16h bf[4];
    const _Float16* bp = BT + (size_t)kc * N + n_base + lm;  // lane%16 = N index, elements = K (table order)
    #pragma unroll
    for (int j = 0; j < 4; ++j)
      #pragma unroll
      for (int v = 0; v < 8; ++v) {
        int k0 = a_k0(v, hsel);
        bf[j][2 * v] = bp[(size_t)k0 * N + j * 16]; bf[j][2 * v + 1] = bp[(size_t)(k0 + 1) * N + j * 16];
      }
    #pragma unroll
    for (int r = 0; r < 2; ++r)
      #pragma unroll
      for (int j = 0; j < 4; ++j)
        acc[r][j] = wmma16(a[r], bf[j], acc[r][j]);
  }

  #pragma unroll
  for (int j = 0; j < 4; ++j) {
    const int col = n_base + j * 16 + lm;
    const float bv = bias[col];
    #pragma unroll
    for (int r = 0; r < 2; ++r) {
      const size_t rb = (size_t)(m_base + r * 16 + 8 * hsel) * N + col;
      #pragma unroll
      for (int v = 0; v < 8; ++v) {
        const size_t idx = rb + (size_t)v * N;
        float val = acc[r][j][v] + bv;
        if (MODE == 0) {
          Ch[idx] = (_Float16)val;
        } else if (MODE == 1) {
          Ch[idx] = (_Float16)fmaxf(val, 0.0f);
        } else {
          Cf[idx] += val;   // residual accumulate in place
        }
      }
    }
  }
}

// ------------------------------------------------------------------
// flash attention: one wave per (event b, head h, 16-query tile)
// qkv f16 rows of width 768: [Q | K | V], each head slice HD=32
// ------------------------------------------------------------------
__global__ __launch_bounds__(32)
void attn_kernel(const _Float16* __restrict__ qkv, const int* __restrict__ valid,
                 _Float16* __restrict__ o16) {
  const int qt = blockIdx.x, h = blockIdx.y, b = blockIdx.z;
  const int lane = threadIdx.x;
  const int lm = lane & 15, hsel = lane >> 4;
  const float scale = 0.17677669529663687f;  // 1/sqrt(32)
  const int QKVW = 3 * D_MODEL;
  __shared__ _Float16 P[16][32];

  // Q A-fragment (16 queries x 32 dims, K = head_dim)
  v16h aQ;
  {
    const _Float16* qp = qkv + (size_t)(b * SMAX + qt * 16 + lm) * QKVW + h * HD;
    #pragma unroll
    for (int v = 0; v < 8; ++v) {
      h2 p = *(const h2*)(qp + a_k0(v, hsel));
      aQ[2 * v] = p.x; aQ[2 * v + 1] = p.y;
    }
  }

  float mrun[8], lrun[8];
  v8f o0 = (v8f){}, o1 = (v8f){};
  #pragma unroll
  for (int v = 0; v < 8; ++v) { mrun[v] = -1e30f; lrun[v] = 0.0f; }

  for (int kc = 0; kc < SMAX; kc += 32) {
    // B fragments for scores: lane%16 = key, elements = dim (table order)
    v16h bK0, bK1;
    {
      const _Float16* kp = qkv + (size_t)(b * SMAX + kc + lm) * QKVW + D_MODEL + h * HD;
      #pragma unroll
      for (int v = 0; v < 8; ++v) {
        int k0 = a_k0(v, hsel);
        h2 p = *(const h2*)(kp + k0);
        h2 q = *(const h2*)(kp + (size_t)16 * QKVW + k0);
        bK0[2 * v] = p.x; bK0[2 * v + 1] = p.y;
        bK1[2 * v] = q.x; bK1[2 * v + 1] = q.y;
      }
    }
    v8f z = (v8f){};
    v8f s0 = wmma16(aQ, bK0, z);
    v8f s1 = wmma16(aQ, bK1, z);

    int va = valid[b * SMAX + kc + lm];
    int vb = valid[b * SMAX + kc + 16 + lm];
    float m_new[8], p0[8], p1[8];
    #pragma unroll
    for (int v = 0; v < 8; ++v) {
      float x0 = va ? s0[v] * scale : -1e9f;
      float x1 = vb ? s1[v] * scale : -1e9f;
      s0[v] = x0; s1[v] = x1;
      float t = fmaxf(x0, x1);              // row reduce across 16 lanes of this half
      t = fmaxf(t, __shfl_xor(t, 1, 32));
      t = fmaxf(t, __shfl_xor(t, 2, 32));
      t = fmaxf(t, __shfl_xor(t, 4, 32));
      t = fmaxf(t, __shfl_xor(t, 8, 32));
      m_new[v] = fmaxf(mrun[v], t);
    }
    #pragma unroll
    for (int v = 0; v < 8; ++v) {
      p0[v] = __expf(s0[v] - m_new[v]);
      p1[v] = __expf(s1[v] - m_new[v]);
      float r = p0[v] + p1[v];
      r += __shfl_xor(r, 1, 32);
      r += __shfl_xor(r, 2, 32);
      r += __shfl_xor(r, 4, 32);
      r += __shfl_xor(r, 8, 32);
      float corr = __expf(mrun[v] - m_new[v]);
      lrun[v] = lrun[v] * corr + r;
      o0[v] *= corr; o1[v] *= corr;
      mrun[v] = m_new[v];
    }
    // re-layout P (C/D layout) -> A-fragment layout via LDS
    __syncthreads();
    #pragma unroll
    for (int v = 0; v < 8; ++v) {
      P[v + 8 * hsel][lm]      = (_Float16)p0[v];
      P[v + 8 * hsel][16 + lm] = (_Float16)p1[v];
    }
    __syncthreads();
    v16h aP;
    #pragma unroll
    for (int v = 0; v < 8; ++v) {
      int k0 = a_k0(v, hsel);
      aP[2 * v] = P[lm][k0]; aP[2 * v + 1] = P[lm][k0 + 1];
    }
    // B fragments for V: lane%16 = dim, elements = key (table order)
    const _Float16* vp = qkv + (size_t)(b * SMAX + kc) * QKVW + 2 * D_MODEL + h * HD + lm;
    v16h bV0, bV1;
    #pragma unroll
    for (int v = 0; v < 8; ++v) {
      int k0 = a_k0(v, hsel);
      bV0[2 * v] = vp[(size_t)k0 * QKVW]; bV0[2 * v + 1] = vp[(size_t)(k0 + 1) * QKVW];
      bV1[2 * v] = vp[(size_t)k0 * QKVW + 16]; bV1[2 * v + 1] = vp[(size_t)(k0 + 1) * QKVW + 16];
    }
    o0 = wmma16(aP, bV0, o0);
    o1 = wmma16(aP, bV1, o1);
  }

  #pragma unroll
  for (int v = 0; v < 8; ++v) {
    int row = b * SMAX + qt * 16 + v + 8 * hsel;
    float inv = 1.0f / lrun[v];
    o16[(size_t)row * D_MODEL + h * HD + lm]      = (_Float16)(o0[v] * inv);
    o16[(size_t)row * D_MODEL + h * HD + 16 + lm] = (_Float16)(o1[v] * inv);
  }
}

// ------------------------------------------------------------------
// masked mean pooling
// ------------------------------------------------------------------
__global__ __launch_bounds__(256)
void pool_kernel(const float* __restrict__ x, const int* __restrict__ valid,
                 float* __restrict__ embOut) {
  int b = blockIdx.x, d = threadIdx.x;
  float s = 0.0f; int cnt = 0;
  for (int j = 0; j < SMAX; ++j) {
    int v = valid[b * SMAX + j];
    cnt += v;
    if (v) s += x[(size_t)(b * SMAX + j) * D_MODEL + d];
  }
  float denom = (float)(cnt > 0 ? cnt : 1);
  embOut[b * D_MODEL + d] = s / denom;
}

// ------------------------------------------------------------------
// head: relu(emb @ W1^T + b1) @ W2^T + b2, then L2 normalize
// ------------------------------------------------------------------
__global__ __launch_bounds__(256)
void head_kernel(const float* __restrict__ emb,
                 const float* __restrict__ w1, const float* __restrict__ b1,
                 const float* __restrict__ w2, const float* __restrict__ b2,
                 float* __restrict__ out) {
  int b = blockIdx.x, n = threadIdx.x;
  __shared__ float e_sh[256], t_sh[256], vv[3];
  e_sh[n] = emb[b * 256 + n]; __syncthreads();
  float acc = b1[n];
  const float* wr = w1 + (size_t)n * 256;
  #pragma unroll 4
  for (int k = 0; k < 256; ++k) acc += wr[k] * e_sh[k];
  t_sh[n] = fmaxf(acc, 0.0f); __syncthreads();
  if (n < 3) {
    float a = b2[n];
    const float* w = w2 + n * 256;
    for (int k = 0; k < 256; ++k) a += w[k] * t_sh[k];
    vv[n] = a;
  }
  __syncthreads();
  if (n == 0) {
    float nm = sqrtf(vv[0] * vv[0] + vv[1] * vv[1] + vv[2] * vv[2]) + 1e-8f;
    out[b * 3 + 0] = vv[0] / nm;
    out[b * 3 + 1] = vv[1] / nm;
    out[b * 3 + 2] = vv[2] / nm;
  }
}

// ------------------------------------------------------------------
extern "C" void kernel_launch(void* const* d_in, const int* in_sizes, int n_in,
                              void* d_out, int out_size, void* d_ws, size_t ws_size,
                              hipStream_t stream) {
  const float* dom_emb  = (const float*)d_in[0];
  const float* geometry = (const float*)d_in[1];
  const float* geo_w1   = (const float*)d_in[2];
  const float* geo_b1   = (const float*)d_in[3];
  const float* geo_w2   = (const float*)d_in[4];
  const float* geo_b2   = (const float*)d_in[5];
  const float* qkv_w    = (const float*)d_in[6];
  const float* qkv_b    = (const float*)d_in[7];
  const float* out_w    = (const float*)d_in[8];
  const float* out_b    = (const float*)d_in[9];
  const float* ln1_w    = (const float*)d_in[10];
  const float* ln1_b    = (const float*)d_in[11];
  const float* ln2_w    = (const float*)d_in[12];
  const float* ln2_b    = (const float*)d_in[13];
  const float* ffn_w1   = (const float*)d_in[14];
  const float* ffn_b1   = (const float*)d_in[15];
  const float* ffn_w2   = (const float*)d_in[16];
  const float* ffn_b2   = (const float*)d_in[17];
  const float* head_w1  = (const float*)d_in[18];
  const float* head_b1  = (const float*)d_in[19];
  const float* head_w2  = (const float*)d_in[20];
  const float* head_b2  = (const float*)d_in[21];
  const int*   dom_ids  = (const int*)d_in[22];
  const int*   ev_idx   = (const int*)d_in[23];
  float* out = (float*)d_out;

  // workspace carve-out (256B aligned)
  char* wp = (char*)d_ws;
  auto alloc = [&](size_t bytes) -> char* {
    char* p = wp; wp += (bytes + 255) & ~(size_t)255; return p;
  };
  int*      counts  = (int*)alloc(BATCH * 4);
  int*      offsets = (int*)alloc(BATCH * 4);
  int*      pos     = (int*)alloc((size_t)T_DOMS * 4);
  int*      valid   = (int*)alloc((size_t)BS * 4);
  float*    geoh    = (float*)alloc((size_t)T_DOMS * 128 * 4);
  float*    x       = (float*)alloc((size_t)BS * D_MODEL * 4);
  _Float16* h16     = (_Float16*)alloc((size_t)BS * D_MODEL * 2);
  _Float16* qkv16   = (_Float16*)alloc((size_t)BS * 3 * D_MODEL * 2);
  _Float16* o16     = (_Float16*)alloc((size_t)BS * D_MODEL * 2);
  _Float16* a16     = (_Float16*)alloc((size_t)BS * DFF * 2);
  float*    embp    = (float*)alloc((size_t)BATCH * D_MODEL * 4);
  _Float16* qkvT    = (_Float16*)alloc((size_t)NLAYER * D_MODEL * 3 * D_MODEL * 2);
  _Float16* outT    = (_Float16*)alloc((size_t)NLAYER * D_MODEL * D_MODEL * 2);
  _Float16* ffn1T   = (_Float16*)alloc((size_t)NLAYER * D_MODEL * DFF * 2);
  _Float16* ffn2T   = (_Float16*)alloc((size_t)NLAYER * DFF * D_MODEL * 2);
  (void)in_sizes; (void)n_in; (void)out_size; (void)ws_size;

  // init
  hipMemsetAsync(counts, 0, BATCH * 4, stream);
  hipMemsetAsync(valid, 0, (size_t)BS * 4, stream);
  hipMemsetAsync(x, 0, (size_t)BS * D_MODEL * 4, stream);

  // weight conversion (f32 NxK -> f16 KxN)
  for (int l = 0; l < NLAYER; ++l) {
    wcast_kernel<<<(3 * D_MODEL * D_MODEL + 255) / 256, 256, 0, stream>>>(
        qkv_w + (size_t)l * 3 * D_MODEL * D_MODEL, qkvT + (size_t)l * D_MODEL * 3 * D_MODEL,
        3 * D_MODEL, D_MODEL);
    wcast_kernel<<<(D_MODEL * D_MODEL + 255) / 256, 256, 0, stream>>>(
        out_w + (size_t)l * D_MODEL * D_MODEL, outT + (size_t)l * D_MODEL * D_MODEL,
        D_MODEL, D_MODEL);
    wcast_kernel<<<(DFF * D_MODEL + 255) / 256, 256, 0, stream>>>(
        ffn_w1 + (size_t)l * DFF * D_MODEL, ffn1T + (size_t)l * D_MODEL * DFF,
        DFF, D_MODEL);
    wcast_kernel<<<(D_MODEL * DFF + 255) / 256, 256, 0, stream>>>(
        ffn_w2 + (size_t)l * D_MODEL * DFF, ffn2T + (size_t)l * DFF * D_MODEL,
        D_MODEL, DFF);
  }

  // ragged packing
  count_kernel<<<(T_DOMS + 255) / 256, 256, 0, stream>>>(ev_idx, counts);
  scan_kernel<<<1, 64, 0, stream>>>(counts, offsets);
  pos_kernel<<<(T_DOMS + 255) / 256, 256, 0, stream>>>(ev_idx, offsets, pos);

  // geometry MLP + scatter into padded x
  geo_hidden_kernel<<<T_DOMS, 128, 0, stream>>>(geometry, dom_ids, geo_w1, geo_b1, geoh);
  geo_scatter_kernel<<<T_DOMS, 256, 0, stream>>>(dom_emb, geoh, geo_w2, geo_b2,
                                                 ev_idx, pos, x, valid);

  const dim3 gemmBlk(128);
  for (int l = 0; l < NLAYER; ++l) {
    // LN1 -> h16
    ln_kernel<<<BS, 256, 0, stream>>>(x, ln1_w + l * D_MODEL, ln1_b + l * D_MODEL, h16);
    // QKV: (BS x 256) x (256 x 768) -> qkv16
    gemm_wmma_kernel<0><<<dim3(BS / 128, (3 * D_MODEL) / 64), gemmBlk, 0, stream>>>(
        h16, qkvT + (size_t)l * D_MODEL * 3 * D_MODEL, qkv_b + l * 3 * D_MODEL,
        nullptr, qkv16, 3 * D_MODEL, D_MODEL);
    // attention -> o16
    attn_kernel<<<dim3(SMAX / 16, NHEAD, BATCH), 32, 0, stream>>>(qkv16, valid, o16);
    // out proj + residual -> x
    gemm_wmma_kernel<2><<<dim3(BS / 128, D_MODEL / 64), gemmBlk, 0, stream>>>(
        o16, outT + (size_t)l * D_MODEL * D_MODEL, out_b + l * D_MODEL,
        x, nullptr, D_MODEL, D_MODEL);
    // LN2 -> h16
    ln_kernel<<<BS, 256, 0, stream>>>(x, ln2_w + l * D_MODEL, ln2_b + l * D_MODEL, h16);
    // FFN1 + relu -> a16
    gemm_wmma_kernel<1><<<dim3(BS / 128, DFF / 64), gemmBlk, 0, stream>>>(
        h16, ffn1T + (size_t)l * D_MODEL * DFF, ffn_b1 + l * DFF,
        nullptr, a16, DFF, D_MODEL);
    // FFN2 + residual -> x
    gemm_wmma_kernel<2><<<dim3(BS / 128, D_MODEL / 64), gemmBlk, 0, stream>>>(
        a16, ffn2T + (size_t)l * DFF * D_MODEL, ffn_b2 + l * D_MODEL,
        x, nullptr, D_MODEL, DFF);
  }

  // pooling + head
  pool_kernel<<<BATCH, 256, 0, stream>>>(x, valid, embp);
  head_kernel<<<BATCH, 256, 0, stream>>>(embp, head_w1, head_b1, head_w2, head_b2, out);
}
